// CausalSelfAttention_64089501991334
// MI455X (gfx1250) — matrix operand order repaired and checked by reading, not compiled, not run on hardware
//
#include <hip/hip_runtime.h>

// ---------------------------------------------------------------------------
// MLA causal attention for MI455X (gfx1250), wave32 + WMMA f16->f32.
// WMMA operands: contiguous 2x ds_load_b128, bank-conflict-free padded pitches.
// K tiles staged with async global->LDS copies, software-pipelined one tile
// ahead (fetch of tile jt+1 overlaps softmax + P@K WMMA of tile jt).
// ---------------------------------------------------------------------------

typedef __attribute__((ext_vector_type(16))) _Float16 v16h;
typedef __attribute__((ext_vector_type(8)))  _Float16 v8h;
typedef __attribute__((ext_vector_type(2)))  _Float16 v2h;
typedef __attribute__((ext_vector_type(8)))  float    v8f;
typedef __attribute__((ext_vector_type(4)))  int      v4i;

#define C_DIM  1024
#define QL_DIM 512
#define KV_DIM 512
#define NH     16
#define HS     64
#define T_SEQ  2048
#define B_SZ   2
#define BT     (B_SZ * T_SEQ)

// padded LDS pitches (in halfs / floats) -> bank-conflict-free b128 loads
#define LDQ   520   // 512-wide f16 rows (qlat, o16, vht)
#define LDK   520   // K tile row-major
#define LDKT  40    // K tile latent-major (32 keys + pad)
#define LDP   40    // P tile (32 cols + pad)
#define LDS_S 33    // S tile f32
#define LDQS  72    // 64-wide f16 rows (qstage, keff block)

// ---------------------------------------------------------------------------
// Async global->LDS copy (CDNA5 GLOBAL_LOAD_ASYNC_TO_LDS_B128, ASYNCcnt),
// with a synchronous fallback at identical program points.
// ---------------------------------------------------------------------------
#if defined(__gfx1250__) &&                                        \
    __has_builtin(__builtin_amdgcn_global_load_async_to_lds_b128) && \
    __has_builtin(__builtin_amdgcn_s_wait_asynccnt)
#define HAVE_ASYNC_LDS 1
#else
#define HAVE_ASYNC_LDS 0
#endif

#if HAVE_ASYNC_LDS
typedef __attribute__((address_space(1))) v4i v4i_g;   // global int4
typedef __attribute__((address_space(3))) v4i v4i_l;   // LDS int4
#endif

__device__ __forceinline__ void cp16_g2l(void* lds, const void* g)
{
#if HAVE_ASYNC_LDS
    __builtin_amdgcn_global_load_async_to_lds_b128(
        (v4i_g*)(unsigned long long)g,
        (v4i_l*)(unsigned int)(unsigned long long)lds,
        0, 0);
#else
    *(uint4*)lds = *(const uint4*)g;
#endif
}

__device__ __forceinline__ void cp_wait()
{
#if HAVE_ASYNC_LDS
    __builtin_amdgcn_s_wait_asynccnt(0);
#endif
}

// ============================================================================
// Generic fp32 tiled GEMM with arbitrary strides (prep stage, accuracy path):
//   out[i,j] = scale * sum_k A[i*as_i + k*as_k] * B[k*bs_k + j*bs_j]
// ============================================================================
__global__ void __launch_bounds__(256)
gemm_f32(const float* __restrict__ A, const float* __restrict__ B,
         float* __restrict__ outF, _Float16* __restrict__ outH,
         int N, int K,
         long as_i, long as_k, long bs_k, long bs_j,
         long oh_i, long oh_j, float scale)
{
    __shared__ float As[16][17];
    __shared__ float Bs[16][17];
    const int tx = threadIdx.x, ty = threadIdx.y;
    const long i = (long)blockIdx.y * 16 + ty;
    const long j = (long)blockIdx.x * 16 + tx;
    float acc = 0.f;
    for (int k0 = 0; k0 < K; k0 += 16) {
        As[ty][tx] = A[i * as_i + (long)(k0 + tx) * as_k];
        Bs[ty][tx] = B[(long)(k0 + ty) * bs_k + j * bs_j];
        __syncthreads();
#pragma unroll
        for (int kk = 0; kk < 16; ++kk) acc = fmaf(As[ty][kk], Bs[kk][tx], acc);
        __syncthreads();
    }
    acc *= scale;
    if (outF) outF[i * (long)N + j] = acc;
    if (outH) outH[i * oh_i + j * oh_j] = (_Float16)acc;
}

// ============================================================================
// WMMA fragment helpers (contiguous k = 16*(lane/16) + s in both A and B, wave32)
// ============================================================================
__device__ __forceinline__ v16h cat16(v8h a, v8h b)
{
    return __builtin_shufflevector(a, b, 0, 1, 2, 3, 4, 5, 6, 7,
                                         8, 9, 10, 11, 12, 13, 14, 15);
}

// A fragment (16x32 f16) from row-major slab.
__device__ __forceinline__ v16h load_a_frag(const _Float16* base, int ld,
                                            int row0, int k0, int lane)
{
    const _Float16* p = base + (long)(row0 + (lane & 15)) * ld + k0 + ((lane >> 4) << 4);
    return cat16(*(const v8h*)p, *(const v8h*)(p + 8));
}

// B fragment (32x16 f16), K-major storage: B[k,n] = base[(col0+n)*ld + k0+k].
__device__ __forceinline__ v16h load_b_frag_nk(const _Float16* base, int ld,
                                               int col0, int k0, int lane)
{
    const _Float16* p = base + (long)(col0 + (lane & 15)) * ld + k0 + ((lane >> 4) << 4);
    return cat16(*(const v8h*)p, *(const v8h*)(p + 8));
}

__device__ __forceinline__ v8f wmma16(v16h a, v16h b, v8f c)
{
    return __builtin_amdgcn_wmma_f32_16x16x32_f16(false, a, false, b,
                                                  (short)0, c, false, false);
}

// ============================================================================
// Flash attention in latent space. One WG = 64 q rows of one (b, head).
// 256 threads = 8 wave32s: rg = wave&3 (16-row strip), chalf = wave>>2.
// ============================================================================
// LDS map (bytes):
//   qlat  [64][520] f16 @ 0       (66560)   | o16 overlay after j-loop
//   ktile [32][520] f16 @ 66560   (33280)   | kblk [512][72] overlay pre-loop
//   ktileT[512][40] f16 @ 99840   (40960)   | vht [64][520] overlay after loop
//   sbuf  [64][33]  f32 @ 140800  (8448)    | qstage [64][72] overlay pre-loop
//   pbuf  [64][40]  f16 @ 150016  (5120)
//   stats m/l/a[64] f32 @ 155136  (768)     -> total 155904
#define SMEM_BYTES 155904

__global__ void __launch_bounds__(256)
mla_flash_kernel(const _Float16* __restrict__ q16,    // (BT, C)   f16
                 const _Float16* __restrict__ kv16,   // (BT, KV)  f16
                 const _Float16* __restrict__ keffT,  // (KV, C)   f16, latent-major, *1/8
                 const float*    __restrict__ veff,   // (KV, C)   f32
                 float* __restrict__ y)               // (BT, C)   f32
{
    extern __shared__ char smem[];
    _Float16* qlat   = (_Float16*)(smem);             // [64][LDQ]
    _Float16* ktile  = (_Float16*)(smem + 66560);     // [32][LDK]
    _Float16* ktileT = (_Float16*)(smem + 99840);     // [512][LDKT]
    float*    sbuf   = (float*)   (smem + 140800);    // [64][LDS_S]
    _Float16* pbuf   = (_Float16*)(smem + 150016);    // [64][LDP]
    float*    mrow   = (float*)   (smem + 155136);
    float*    lrow   = mrow + 64;
    float*    arow   = lrow + 64;
    // overlays
    _Float16* kblk   = ktile;                         // [512][LDQS] keff head blk
    _Float16* qstage = (_Float16*)(smem + 140800);    // [64][LDQS]
    _Float16* o16    = qlat;                          // [64][LDQ]
    _Float16* vht    = ktile;                         // [64][LDQ] (d-major v_eff)

    const int qt = blockIdx.x, h = blockIdx.y, b = blockIdx.z;
    const int tid  = threadIdx.x;
    const int lane = tid & 31, wave = tid >> 5;
    const int rg    = wave & 3;     // 16-row strip of the 64 q rows
    const int chalf = wave >> 2;    // latent half (256 cols) for O
    const int qbase = qt * 64;
    const long row0 = (long)b * T_SEQ + qbase;

    // async K-tile stage: 32 rows x 512 halfs -> ktile (row-major, padded)
    auto stage_k = [&](int jbase) {
        for (int i = tid; i < 32 * 64; i += 256) {
            int r = i >> 6, c = i & 63;
            cp16_g2l(ktile + r * LDK + c * 8,
                     kv16 + ((long)b * T_SEQ + jbase + r) * KV_DIM + c * 8);
        }
    };

    // ---- stage q tile (head h) and keff head block (latent-major), async ----
    for (int i = tid; i < 64 * 8; i += 256) {
        int r = i >> 3, c = i & 7;
        cp16_g2l(qstage + r * LDQS + c * 8,
                 q16 + (row0 + r) * C_DIM + h * HS + c * 8);
    }
    for (int i = tid; i < 512 * 8; i += 256) {      // kblk[l][d] = keffT[l][h*64+d]
        int l = i >> 3, c = i & 7;
        cp16_g2l(kblk + l * LDQS + c * 8,
                 keffT + (long)l * C_DIM + h * HS + c * 8);
    }
    if (tid < 64) { mrow[tid] = -1e30f; lrow[tid] = 0.f; }
    cp_wait();
    __syncthreads();

    // ---- Q_lat = qstage(64x64) @ keff_blk(64x512) -> qlat f16 (row-major) ----
#pragma unroll
    for (int ct = 0; ct < 16; ++ct) {
        const int lc = chalf * 256 + ct * 16;
        v8f acc = {};
#pragma unroll
        for (int kc = 0; kc < 2; ++kc) {
            v16h a  = load_a_frag(qstage, LDQS, rg * 16, kc * 32, lane);
            v16h bf = load_b_frag_nk(kblk, LDQS, lc, kc * 32, lane);
            acc = wmma16(a, bf, acc);
        }
        const int n  = lc + (lane & 15);
        const int r0 = rg * 16 + ((lane >> 4) << 3);
#pragma unroll
        for (int r = 0; r < 8; ++r)
            qlat[(long)(r0 + r) * LDQ + n] = (_Float16)acc[r];
    }
    __syncthreads();   // kblk reads done -> ktile region reusable

    // ---- O accumulators: rows rg*16..+15, latent cols chalf*256..+255 ----
    v8f o[16] = {};

    // ---- software-pipelined flash loop over 32-row key tiles (causal) ----
    const int jt_max = 2 * qt + 1;
    stage_k(0);                           // prefetch tile 0
    for (int jt = 0; jt <= jt_max; ++jt) {
        const int jbase = jt * 32;
        cp_wait();                        // this wave's async stores landed
        __syncthreads();                  // all waves: ktile ready, ktileT free
        // transpose K -> ktileT[l][n] (2x8 blocks, v8h reads, v2h writes)
        for (int i = tid; i < 16 * 64; i += 256) {
            int lp = i >> 4, np = i & 15;                   // l block, n pair
            v8h r0v = *(const v8h*)(ktile + (2 * np) * LDK + lp * 8);
            v8h r1v = *(const v8h*)(ktile + (2 * np + 1) * LDK + lp * 8);
#pragma unroll
            for (int e = 0; e < 8; ++e) {
                v2h pr; pr[0] = r0v[e]; pr[1] = r1v[e];
                *(v2h*)(ktileT + (lp * 8 + e) * LDKT + 2 * np) = pr;
            }
        }
        __syncthreads();

        // S = Q_lat @ K^T  (64x32): wave -> (rg rows, cs = wave>>2 -> 16 cols)
        {
            const int cs = wave >> 2;
            v8f s0 = {};
            for (int kk = 0; kk < 16; ++kk) {
                v16h a  = load_a_frag(qlat, LDQ, rg * 16, kk * 32, lane);
                v16h bf = load_b_frag_nk(ktile, LDK, cs * 16, kk * 32, lane);
                s0 = wmma16(a, bf, s0);
            }
            const int n0 = cs * 16 + (lane & 15);
            const int r0 = rg * 16 + ((lane >> 4) << 3);
#pragma unroll
            for (int r = 0; r < 8; ++r)
                sbuf[(r0 + r) * LDS_S + n0] = s0[r];
        }
        __syncthreads();                  // sbuf ready; ktile now dead

        // prefetch next K tile while softmax + P@K run (writes only ktile)
        if (jt < jt_max) stage_k(jbase + 32);

        // online softmax, one thread per row
        if (tid < 64) {
            const int m = tid;
            int nv = qbase + m - jbase + 1;                 // causal valid count
            nv = nv < 0 ? 0 : (nv > 32 ? 32 : nv);
            const float* srow = sbuf + m * LDS_S;
            float mx = mrow[m];
            for (int n = 0; n < nv; ++n) mx = fmaxf(mx, srow[n]);
            const float alpha = __expf(mrow[m] - mx);
            float sum = lrow[m] * alpha;
            _Float16* prow = pbuf + m * LDP;
            for (int n = 0; n < 32; ++n) {
                const float p = (n < nv) ? __expf(srow[n] - mx) : 0.f;
                sum += p;
                prow[n] = (_Float16)p;
            }
            mrow[m] = mx; lrow[m] = sum; arow[m] = alpha;
        }
        __syncthreads();

        // rescale O then O += P(64x32) @ K(32x512)
        {
            const int r0 = rg * 16 + ((lane >> 4) << 3);
            float av[8];
#pragma unroll
            for (int r = 0; r < 8; ++r) av[r] = arow[r0 + r];
#pragma unroll
            for (int ct = 0; ct < 16; ++ct)
#pragma unroll
                for (int r = 0; r < 8; ++r) o[ct][r] *= av[r];

            v16h a = load_a_frag(pbuf, LDP, rg * 16, 0, lane);
#pragma unroll
            for (int ct = 0; ct < 16; ++ct) {
                v16h bf = load_b_frag_nk(ktileT, LDKT, chalf * 256 + ct * 16, 0, lane);
                o[ct] = wmma16(a, bf, o[ct]);
            }
        }
    }
    __syncthreads();   // done with qlat/ktile/ktileT roles

    // ---- finalize: o16 = O / l (f16, row-major) ; stage v_eff head blk d-major ----
    {
        const int r0 = rg * 16 + ((lane >> 4) << 3);
        float linv[8];
#pragma unroll
        for (int r = 0; r < 8; ++r) linv[r] = 1.f / lrow[r0 + r];
#pragma unroll
        for (int ct = 0; ct < 16; ++ct) {
            const int n = chalf * 256 + ct * 16 + (lane & 15);
#pragma unroll
            for (int r = 0; r < 8; ++r)
                o16[(long)(r0 + r) * LDQ + n] = (_Float16)(o[ct][r] * linv[r]);
        }
        for (int i = tid; i < KV_DIM * HS; i += 256) {      // vht[d][l]
            int l = i >> 6, d = i & 63;
            vht[(long)d * LDQ + l] = (_Float16)veff[(long)l * C_DIM + h * HS + d];
        }
    }
    __syncthreads();

    // ---- y tile = o16(64x512) @ v_eff(512x64): wave -> (rg rows, 32 cols) ----
    {
        const int cs = wave >> 2;   // cols 32*cs
        v8f y0 = {}, y1 = {};
        for (int kk = 0; kk < 16; ++kk) {
            v16h a  = load_a_frag(o16, LDQ, rg * 16, kk * 32, lane);
            v16h b0 = load_b_frag_nk(vht, LDQ, cs * 32,      kk * 32, lane);
            v16h b1 = load_b_frag_nk(vht, LDQ, cs * 32 + 16, kk * 32, lane);
            y0 = wmma16(a, b0, y0);
            y1 = wmma16(a, b1, y1);
        }
        const int n0 = cs * 32 + (lane & 15);
        const int r0 = rg * 16 + ((lane >> 4) << 3);
#pragma unroll
        for (int r = 0; r < 8; ++r) {
            y[(row0 + r0 + r) * C_DIM + h * HS + n0]      = y0[r];
            y[(row0 + r0 + r) * C_DIM + h * HS + n0 + 16] = y1[r];
        }
    }
}

// ============================================================================
// Host launcher
// ============================================================================
extern "C" void kernel_launch(void* const* d_in, const int* in_sizes, int n_in,
                              void* d_out, int out_size, void* d_ws, size_t ws_size,
                              hipStream_t stream)
{
    const float* x    = (const float*)d_in[0];
    const float* Wdq  = (const float*)d_in[1];   // (QL, C)
    const float* Wuq  = (const float*)d_in[2];   // (C, QL)
    const float* Wdkv = (const float*)d_in[3];   // (KV, C)
    const float* Wuk  = (const float*)d_in[4];   // (C, KV)
    const float* Wuv  = (const float*)d_in[5];   // (C, KV)
    const float* Wo   = (const float*)d_in[6];   // (C, C)

    float* y_out   = (float*)d_out;                       // (BT, C)
    float* ckv_out = y_out + (long)BT * C_DIM;            // (BT, KV)

    char* ws = (char*)d_ws;
    const size_t MB = 1u << 20;
    _Float16* keffT  = (_Float16*)(ws);                   //  1 MB  (KV x C), latent-major
    _Float16* ckv16  = (_Float16*)(ws + 1 * MB);          //  4 MB  (BT x KV)
    _Float16* q16    = (_Float16*)(ws + 5 * MB);          //  8 MB  (BT x C)
    float*    veff   = (float*)   (ws + 13 * MB);         //  2 MB  (KV x C)
    float*    M1     = (float*)   (ws + 15 * MB);         //  4 MB  (C x C), dead after G2
    float*    cq     = (float*)   (ws + 15 * MB);         //  8 MB  (BT x QL), reuses M1
    // total workspace: 23 MB

    dim3 blk(16, 16);

    // G1: M1 = Wdq^T @ Wuq^T              (C x C, K = QL)
    gemm_f32<<<dim3(C_DIM / 16, C_DIM / 16), blk, 0, stream>>>(
        Wdq, Wuq, M1, nullptr, C_DIM, QL_DIM,
        1, C_DIM, 1, QL_DIM, 0, 0, 1.0f);

    // G2: keffT[l][c] = (M1 @ Wuk)[c][l] / sqrt(hs)   (f16, transposed store)
    gemm_f32<<<dim3(KV_DIM / 16, C_DIM / 16), blk, 0, stream>>>(
        M1, Wuk, nullptr, keffT, KV_DIM, C_DIM,
        C_DIM, 1, KV_DIM, 1, /*oh_i*/ 1, /*oh_j*/ C_DIM, 0.125f);

    // G3: veff = Wuv^T @ Wo^T             (KV x C, K = C), f32
    gemm_f32<<<dim3(C_DIM / 16, KV_DIM / 16), blk, 0, stream>>>(
        Wuv, Wo, veff, nullptr, C_DIM, C_DIM,
        1, KV_DIM, 1, C_DIM, 0, 0, 1.0f);

    // G4: c_kv = x @ Wdkv^T               (BT x KV, K = C), f32 out + f16 copy
    gemm_f32<<<dim3(KV_DIM / 16, BT / 16), blk, 0, stream>>>(
        x, Wdkv, ckv_out, ckv16, KV_DIM, C_DIM,
        C_DIM, 1, 1, C_DIM, KV_DIM, 1, 1.0f);

    // G5: cq = x @ Wdq^T                  (BT x QL, K = C), f32
    gemm_f32<<<dim3(QL_DIM / 16, BT / 16), blk, 0, stream>>>(
        x, Wdq, cq, nullptr, QL_DIM, C_DIM,
        C_DIM, 1, 1, C_DIM, 0, 0, 1.0f);

    // G6: q16 = cq @ Wuq^T                (BT x C, K = QL), f16
    gemm_f32<<<dim3(C_DIM / 16, BT / 16), blk, 0, stream>>>(
        cq, Wuq, nullptr, q16, C_DIM, QL_DIM,
        QL_DIM, 1, 1, QL_DIM, C_DIM, 1, 1.0f);

    // Flash attention: one WG per (64-row q tile, head, batch)
    dim3 grid(T_SEQ / 64, NH, B_SZ);
    mla_flash_kernel<<<grid, 256, SMEM_BYTES, stream>>>(
        q16, ckv16, keffT, veff, y_out);
}
